// StructureInferenceNet_61168924229758
// MI455X (gfx1250) — matrix operand order repaired and checked by reading, not compiled, not run on hardware
//
#include <hip/hip_runtime.h>
#include <hip/hip_bf16.h>
#include <stdint.h>

#define N_BOX 128
#define D     4096
#define TG    12288   // 3*D

typedef __attribute__((ext_vector_type(16))) __bf16 v16bf;
typedef __attribute__((ext_vector_type(8)))  float  v8f;

// ---------- helpers ----------
__device__ __forceinline__ unsigned pack_bf2(float lo, float hi) {
    __hip_bfloat162 h = __float22bfloat162_rn(make_float2(lo, hi));  // v_cvt_pk_bf16_f32
    union { __hip_bfloat162 h; unsigned u; } c; c.h = h;
    return c.u;
}
__device__ __forceinline__ float sigmoidf(float x) {
    return 1.0f / (1.0f + __expf(-x));
}

// ---------- PE = relu((masked ofe) @ w_p), 128x128 ----------
__global__ void k_pe(const float* __restrict__ ofe, const float* __restrict__ iou,
                     const float* __restrict__ w_p, float* __restrict__ PE) {
    int idx = blockIdx.x * blockDim.x + threadIdx.x;  // 16384
    if (idx >= N_BOX * N_BOX) return;
    float s = 0.f;
    if (!(iou[idx] > 0.6f)) {
        const float* e = ofe + (size_t)idx * 12;
        #pragma unroll
        for (int k = 0; k < 12; ++k) s += e[k] * w_p[k];
    }
    PE[idx] = fmaxf(s, 0.f);
}

// ---------- gi_o[j] = dot(w_ih_o[j,:], ofs) + b_ih_o[j] (constant across steps) ----------
__global__ void k_gi(const float* __restrict__ W, const float* __restrict__ ofs,
                     const float* __restrict__ bias, float* __restrict__ out) {
    int wave = threadIdx.x >> 5;
    int lane = threadIdx.x & 31;
    int row  = blockIdx.x * 8 + wave;
    const float* wr = W + (size_t)row * D;
    float s = 0.f;
    for (int k = lane * 4; k < D; k += 128) {
        float4 a = *(const float4*)(wr + k);
        float4 b = *(const float4*)(ofs + k);
        s += a.x * b.x + a.y * b.y + a.z * b.z + a.w * b.w;
    }
    #pragma unroll
    for (int off = 16; off; off >>= 1) s += __shfl_xor(s, off, 32);
    if (lane == 0) out[row] = s + bias[row];
}

// ---------- fp32 -> packed bf16 pairs ----------
__global__ void k_cvt_bf16(const float* __restrict__ src, unsigned* __restrict__ dst, int n2) {
    int idx = blockIdx.x * blockDim.x + threadIdx.x;
    if (idx >= n2) return;
    float2 f = *(const float2*)(src + (size_t)idx * 2);
    dst[idx] = pack_bf2(f.x, f.y);
}

// ---------- zero-fill (for split-K atomic accumulation) ----------
__global__ void k_zero(float4* __restrict__ g, int n4) {
    int idx = blockIdx.x * blockDim.x + threadIdx.x;
    if (idx < n4) g[idx] = make_float4(0.f, 0.f, 0.f, 0.f);
}

// ---------- a = hi@wv1, b = hi@wv2 ----------
__global__ void k_ab(const float* __restrict__ h, const float* __restrict__ wv,
                     float* __restrict__ av, float* __restrict__ bv) {
    int wave = threadIdx.x >> 5;
    int lane = threadIdx.x & 31;
    int row  = blockIdx.x * 8 + wave;
    const float* hr = h + (size_t)row * D;
    float sa = 0.f, sb = 0.f;
    for (int k = lane * 4; k < D; k += 128) {
        float4 x  = *(const float4*)(hr + k);
        float4 w1 = *(const float4*)(wv + k);
        float4 w2 = *(const float4*)(wv + D + k);
        sa += x.x * w1.x + x.y * w1.y + x.z * w1.z + x.w * w1.w;
        sb += x.x * w2.x + x.y * w2.y + x.z * w2.z + x.w * w2.w;
    }
    #pragma unroll
    for (int off = 16; off; off >>= 1) {
        sa += __shfl_xor(sa, off, 32);
        sb += __shfl_xor(sb, off, 32);
    }
    if (lane == 0) { av[row] = sa; bv[row] = sb; }
}

// ---------- Z = softmax_row(PE * tanh(a_i + b_j)) ----------
__global__ void k_softmax(const float* __restrict__ PE, const float* __restrict__ av,
                          const float* __restrict__ bv, float* __restrict__ Z) {
    __shared__ float red[4];
    int i = blockIdx.x, j = threadIdx.x;          // 128 threads/row
    float v = PE[i * N_BOX + j] * tanhf(av[i] + bv[j]);
    float m = v;
    #pragma unroll
    for (int off = 16; off; off >>= 1) m = fmaxf(m, __shfl_xor(m, off, 32));
    if ((j & 31) == 0) red[j >> 5] = m;
    __syncthreads();
    m = fmaxf(fmaxf(red[0], red[1]), fmaxf(red[2], red[3]));
    __syncthreads();
    float e = __expf(v - m);
    float s = e;
    #pragma unroll
    for (int off = 16; off; off >>= 1) s += __shfl_xor(s, off, 32);
    if ((j & 31) == 0) red[j >> 5] = s;
    __syncthreads();
    s = red[0] + red[1] + red[2] + red[3];
    Z[i * N_BOX + j] = e / s;
}

// ---------- M[i,k] = max_j Z[i,j]*hi[j,k], written directly as packed bf16 ----------
__global__ void k_M(const float* __restrict__ Z, const float* __restrict__ h,
                    unsigned* __restrict__ Mbf) {
    __shared__ float zs[N_BOX];
    int i = blockIdx.y;
    int k = blockIdx.x * 512 + threadIdx.x * 2;
    if (threadIdx.x < N_BOX) zs[threadIdx.x] = Z[i * N_BOX + threadIdx.x];
    __syncthreads();
    float m0 = -3.402823466e38f, m1 = -3.402823466e38f;
    #pragma unroll 4
    for (int j = 0; j < N_BOX; ++j) {
        float z = zs[j];
        float2 hv = *(const float2*)(h + (size_t)j * D + k);
        m0 = fmaxf(m0, z * hv.x);
        m1 = fmaxf(m1, z * hv.y);
    }
    Mbf[((size_t)i * D + k) >> 1] = pack_bf2(m0, m1);
}

// ---------- batched split-K GEMM: C[y][128,TG] += X[y] @ W[y]^T over K-split z ----------
// A tile: async-copied to double-buffered LDS (shared by 8 waves).
// B frag: per-lane direct global fp32 loads -> cvt_pk bf16 -> WMMA (no cross-wave reuse).
#define KC     32
#define LDA    20        // dwords per LDS A row (16 pairs + 4 pad; rows 16B-aligned)
#define KSPLIT 2

__global__ void __launch_bounds__(256)
k_gemm3(const unsigned* __restrict__ xh, const unsigned* __restrict__ xm,
        const float* __restrict__ w0, const float* __restrict__ w1,
        const float* __restrict__ w2,
        float* __restrict__ g0, float* __restrict__ g1, float* __restrict__ g2) {
    __shared__ unsigned As[2][128 * LDA];
    const int which = blockIdx.y;
    const unsigned* __restrict__ Xbf = (which == 1) ? xm : xh;
    const float*    __restrict__ W   = (which == 0) ? w0 : ((which == 1) ? w1 : w2);
    float*          __restrict__ C   = (which == 0) ? g0 : ((which == 1) ? g1 : g2);

    const int tid  = threadIdx.x;
    const int wave = tid >> 5, lane = tid & 31;
    const int colbase = blockIdx.x * 128;
    const int wcol    = colbase + wave * 16;

    const int NCH  = D / KC / KSPLIT;              // 64 chunks per split
    const int ch0  = blockIdx.z * NCH;             // first (global) chunk of this split

    v8f acc[8];
    #pragma unroll
    for (int m = 0; m < 8; ++m) acc[m] = (v8f){0.f,0.f,0.f,0.f,0.f,0.f,0.f,0.f};

    // ---- A staging (async copy, 32B per thread per chunk) ----
    const int srow  = tid >> 1;   // 0..127
    const int spart = tid & 1;
    const unsigned* agbase = Xbf + (size_t)srow * (D / 2) + spart * 8;
    unsigned ldsA[2];
    ldsA[0] = (unsigned)(size_t)&As[0][srow * LDA + spart * 8];
    ldsA[1] = (unsigned)(size_t)&As[1][srow * LDA + spart * 8];
    auto issueA = [&](int gchunk, int buf) {
        unsigned long long ga = (unsigned long long)(size_t)(agbase + gchunk * (KC / 2));
        unsigned l = ldsA[buf];
        asm volatile("global_load_async_to_lds_b128 %0, %1, off"
                     :: "v"(l), "v"(ga) : "memory");
        asm volatile("global_load_async_to_lds_b128 %0, %1, off"
                     :: "v"(l + 16u), "v"(ga + 16ull) : "memory");
    };

    // ---- B per-lane direct path: lane owns (col = wcol + lane%16, K half = (lane/16)*16) ----
    const float* wlane = W + (size_t)(wcol + (lane & 15)) * D + ((lane >> 4) << 4);
    float4 br[4], bn[4];
    auto loadB = [&](int gchunk, float4* b) {
        const float* wp = wlane + (size_t)gchunk * KC;
        b[0] = *(const float4*)(wp);
        b[1] = *(const float4*)(wp + 4);
        b[2] = *(const float4*)(wp + 8);
        b[3] = *(const float4*)(wp + 12);
        __builtin_prefetch(wp + KC, 0, 3);
    };

    // prologue: fill A buffer 0 and first B regs
    issueA(ch0, 0);
    loadB(ch0, br);
    asm volatile("s_wait_asynccnt 0x0" ::: "memory");
    __syncthreads();

    for (int ch = 0; ch < NCH; ++ch) {
        const int cur = ch & 1, nxt = cur ^ 1;
        if (ch + 1 < NCH) {           // all waves passed the barrier ending ch-1,
            issueA(ch0 + ch + 1, nxt);// which was the last reader of buffer `nxt`
            loadB(ch0 + ch + 1, bn);
        }

        // B fragment: 16 contiguous fp32 -> 8 packed bf16 pairs
        union { v16bf v; unsigned u[8]; } bfr;
        #pragma unroll
        for (int q = 0; q < 4; ++q) {
            bfr.u[2 * q]     = pack_bf2(br[q].x, br[q].y);
            bfr.u[2 * q + 1] = pack_bf2(br[q].z, br[q].w);
        }
        // A fragments (this code's contiguous k order, same as B): lanes 0-15 K{0..15}, lanes 16-31 K{16..31}
        union { v16bf v; unsigned u[8]; } afr[8];
        #pragma unroll
        for (int m = 0; m < 8; ++m) {
            const unsigned* apf = &As[cur][(m * 16 + (lane & 15)) * LDA + ((lane >> 4) << 3)];
            #pragma unroll
            for (int q = 0; q < 4; ++q) { afr[m].u[q] = apf[q]; afr[m].u[4 + q] = apf[4 + q]; }
        }
        #pragma unroll
        for (int m = 0; m < 8; ++m)
            acc[m] = __builtin_amdgcn_wmma_f32_16x16x32_bf16(
                false, afr[m].v, false, bfr.v, (short)0, acc[m], false, false);

        #pragma unroll
        for (int q = 0; q < 4; ++q) br[q] = bn[q];

        if (ch + 1 < NCH) {
            asm volatile("s_wait_asynccnt 0x0" ::: "memory");
            __syncthreads();
        }
    }

    // epilogue: split-K accumulate. C VGPR layout: VGPR r -> lanes 0-15 M=r, lanes 16-31 M=r+8
    #pragma unroll
    for (int m = 0; m < 8; ++m) {
        int rowb = m * 16 + (lane >> 4) * 8;
        int col  = wcol + (lane & 15);
        float* cp = C + (size_t)rowb * TG + col;
        #pragma unroll
        for (int r = 0; r < 8; ++r) atomicAdd(cp + (size_t)r * TG, acc[m][r]);
    }
}

// ---------- fused GRU-pair combine (+ optional final relu) ----------
__global__ void k_gru(const float* __restrict__ G0,   // hi @ w_hh_o.T
                      const float* __restrict__ G1,   // M  @ w_ih_e.T
                      const float* __restrict__ G2,   // hi @ w_hh_e.T
                      const float* __restrict__ gio,  // ofs @ w_ih_o.T + b_ih_o (row-broadcast)
                      const float* __restrict__ b_hh_o,
                      const float* __restrict__ b_ih_e,
                      const float* __restrict__ b_hh_e,
                      const float* __restrict__ hcur,
                      float* __restrict__ hdst, int do_relu) {
    int idx = blockIdx.x * blockDim.x + threadIdx.x;
    if (idx >= N_BOX * D) return;
    int i = idx >> 12;
    int c = idx & (D - 1);
    size_t gb = (size_t)i * TG + c;
    float h = hcur[idx];

    float go_r = G0[gb]         + b_hh_o[c];
    float go_z = G0[gb + D]     + b_hh_o[c + D];
    float go_n = G0[gb + 2 * D] + b_hh_o[c + 2 * D];
    float r_o = sigmoidf(gio[c] + go_r);
    float z_o = sigmoidf(gio[c + D] + go_z);
    float n_o = tanhf(gio[c + 2 * D] + r_o * go_n);
    float h1 = (1.f - z_o) * n_o + z_o * h;

    float gi_r = G1[gb]         + b_ih_e[c];
    float gi_z = G1[gb + D]     + b_ih_e[c + D];
    float gi_n = G1[gb + 2 * D] + b_ih_e[c + 2 * D];
    float ge_r = G2[gb]         + b_hh_e[c];
    float ge_z = G2[gb + D]     + b_hh_e[c + D];
    float ge_n = G2[gb + 2 * D] + b_hh_e[c + 2 * D];
    float r_e = sigmoidf(gi_r + ge_r);
    float z_e = sigmoidf(gi_z + ge_z);
    float n_e = tanhf(gi_n + r_e * ge_n);
    float h2 = (1.f - z_e) * n_e + z_e * h;

    float o = 0.5f * (h1 + h2);
    if (do_relu) o = fmaxf(o, 0.f);
    hdst[idx] = o;
}

extern "C" void kernel_launch(void* const* d_in, const int* in_sizes, int n_in,
                              void* d_out, int out_size, void* d_ws, size_t ws_size,
                              hipStream_t stream) {
    const float* ofe    = (const float*)d_in[0];
    const float* ofo    = (const float*)d_in[1];
    const float* ofs    = (const float*)d_in[2];
    /* d_in[3] = n_boxes (fixed 128) */
    const float* iou    = (const float*)d_in[4];
    const float* w_p    = (const float*)d_in[5];
    const float* w_v    = (const float*)d_in[6];
    const float* w_ih_o = (const float*)d_in[7];
    const float* w_hh_o = (const float*)d_in[8];
    const float* b_ih_o = (const float*)d_in[9];
    const float* b_hh_o = (const float*)d_in[10];
    const float* w_ih_e = (const float*)d_in[11];
    const float* w_hh_e = (const float*)d_in[12];
    const float* b_ih_e = (const float*)d_in[13];
    const float* b_hh_e = (const float*)d_in[14];
    float* out = (float*)d_out;
    (void)in_sizes; (void)n_in; (void)out_size; (void)ws_size;

    char* p = (char*)d_ws;
    auto carve = [&](size_t bytes) -> char* {
        char* r = p; p += (bytes + 255) & ~(size_t)255; return r;
    };
    float*    PE   = (float*)carve((size_t)N_BOX * N_BOX * 4);
    float*    Z    = (float*)carve((size_t)N_BOX * N_BOX * 4);
    float*    av   = (float*)carve(N_BOX * 4);
    float*    bv   = (float*)carve(N_BOX * 4);
    float*    gio  = (float*)carve((size_t)TG * 4);
    float*    hbuf = (float*)carve((size_t)N_BOX * D * 4);
    unsigned* hbf  = (unsigned*)carve((size_t)N_BOX * D / 2 * 4);
    unsigned* Mbf  = (unsigned*)carve((size_t)N_BOX * D / 2 * 4);
    float*    G0   = (float*)carve((size_t)N_BOX * TG * 4);   // G0..G2 contiguous
    float*    G1   = (float*)carve((size_t)N_BOX * TG * 4);
    float*    G2   = (float*)carve((size_t)N_BOX * TG * 4);

    // step-invariant precomputation
    k_pe<<<(N_BOX * N_BOX + 255) / 256, 256, 0, stream>>>(ofe, iou, w_p, PE);
    k_gi<<<TG / 8, 256, 0, stream>>>(w_ih_o, ofs, b_ih_o, gio);

    const float* hcur = ofo;
    for (int t = 0; t < 2; ++t) {
        k_cvt_bf16<<<(N_BOX * D / 2 + 255) / 256, 256, 0, stream>>>(hcur, hbf, N_BOX * D / 2);
        k_ab<<<N_BOX / 8, 256, 0, stream>>>(hcur, w_v, av, bv);
        k_softmax<<<N_BOX, 128, 0, stream>>>(PE, av, bv, Z);
        dim3 gm(D / 512, N_BOX);
        k_M<<<gm, 256, 0, stream>>>(Z, hcur, Mbf);

        int n4 = 3 * N_BOX * TG / 4;
        k_zero<<<(n4 + 255) / 256, 256, 0, stream>>>((float4*)G0, n4);
        dim3 gg(TG / 128, 3, KSPLIT);
        k_gemm3<<<gg, 256, 0, stream>>>(hbf, Mbf, w_hh_o, w_ih_e, w_hh_e, G0, G1, G2);

        int fin = (t == 1);
        k_gru<<<(N_BOX * D + 255) / 256, 256, 0, stream>>>(
            G0, G1, G2, gio, b_hh_o, b_ih_e, b_hh_e, hcur, fin ? out : hbuf, fin);
        hcur = hbuf;
    }
}
